// PolicyNetwork_26053271617847
// MI455X (gfx1250) — compile-verified, matrix operand order checked by reading, not run on hardware
//
#include <hip/hip_runtime.h>
#include <math.h>

// ---------------------------------------------------------------------------
// Types for CDNA5 WMMA (wave32): v_wmma_f32_16x16x32_bf16
// ---------------------------------------------------------------------------
typedef __bf16 bf16_t;
typedef __bf16 v16bf __attribute__((ext_vector_type(16)));
typedef float  v8f   __attribute__((ext_vector_type(8)));

__device__ __forceinline__ bf16_t f2bf(float f) {
  union { float f; unsigned u; } in; in.f = f;
  unsigned r = in.u + 0x7FFFu + ((in.u >> 16) & 1u);   // round-to-nearest-even
  union { unsigned short s; bf16_t b; } out; out.s = (unsigned short)(r >> 16);
  return out.b;
}

// Load one 16x32 bf16 operand fragment from an LDS row (A: lane=row, B: lane=col).
// Element j of the v16bf holds K = (j>=8 ? 16 : 0) + 8*half + (j&7).
// -> two 16B chunks at byte offsets 16*half and 32+16*half from the row base.
__device__ __forceinline__ v16bf load_frag(const bf16_t* base, int half) {
  union { v16bf v; uint4 q[2]; } u;
  const char* p = (const char*)base + (half << 4);
  u.q[0] = *(const uint4*)(p);
  u.q[1] = *(const uint4*)(p + 32);
  return u.v;
}

__device__ __forceinline__ v8f wmma_bf16(v16bf a, v16bf b, v8f c) {
  return __builtin_amdgcn_wmma_f32_16x16x32_bf16(
      /*neg_a=*/false, a, /*neg_b=*/false, b,
      /*c_mod=*/(short)0, c, /*reuse_a=*/false, /*reuse_b=*/false);
}

__device__ __forceinline__ float apply_act(float v, int act) {
  if (act == 1) return v > 0.f ? v : 0.f;                       // relu
  if (act == 2) return 0.5f * v * (1.f + erff(v * 0.70710678118654752f)); // exact gelu
  return v;
}

// ---------------------------------------------------------------------------
// Generic GEMM: C[M,N] = act(A[M,K] @ W[K,N] + bias[N])
// Block: 256 threads = 8 waves, tile 128(M) x 64(N), K-step 32, bf16 WMMA.
// K must be a multiple of 32 (true for all uses: 256/512/1024/544/256).
// ---------------------------------------------------------------------------
#define AST 40   // LDS row stride (bf16 elems) for A tile (32 + 8 pad)
#define BST 40   // LDS row stride for transposed W tile

__global__ __launch_bounds__(256)
void gemm_bias_act(const float* __restrict__ A, const float* __restrict__ W,
                   const float* __restrict__ bias, float* __restrict__ C,
                   int M, int N, int K, int act)
{
  __shared__ __align__(16) bf16_t As[128 * AST];
  __shared__ __align__(16) bf16_t Bs[64 * BST];

  const int t    = threadIdx.x;
  const int m0   = blockIdx.y * 128;
  const int n0   = blockIdx.x * 64;
  const int wid  = t >> 5;
  const int lane = t & 31;
  const int wm   = wid >> 1;      // 0..3 (M)
  const int wn   = wid & 1;       // 0..1 (N)
  const int half = lane >> 4;
  const int lr   = lane & 15;

  const v8f vz = {0.f, 0.f, 0.f, 0.f, 0.f, 0.f, 0.f, 0.f};
  v8f acc[2][2];
  acc[0][0] = vz; acc[0][1] = vz; acc[1][0] = vz; acc[1][1] = vz;

  for (int k0 = 0; k0 < K; k0 += 32) {
    // --- stage A tile 128x32 (f32 -> bf16), 4 float4 per thread ---
#pragma unroll
    for (int it = 0; it < 4; ++it) {
      int id   = t + it * 256;      // 0..1023
      int row  = id >> 3;
      int quad = id & 7;
      int gm   = m0 + row;
      float4 v = make_float4(0.f, 0.f, 0.f, 0.f);
      if (gm < M) v = *(const float4*)(A + (size_t)gm * K + k0 + quad * 4);
      bf16_t p4[4] = {f2bf(v.x), f2bf(v.y), f2bf(v.z), f2bf(v.w)};
      *(uint2*)&As[row * AST + quad * 4] = *(const uint2*)p4;
    }
    // --- stage W tile 32x64 transposed -> Bs[n][k] ---
#pragma unroll
    for (int it = 0; it < 2; ++it) {
      int id = t + it * 256;        // 0..511
      int kr = id >> 4;             // 0..31
      int nq = (id & 15) * 4;       // 0,4,...,60
      const float* wr = W + (size_t)(k0 + kr) * N;
#pragma unroll
      for (int j = 0; j < 4; ++j) {
        int gn = n0 + nq + j;
        float wv = (gn < N) ? wr[gn] : 0.f;
        Bs[(nq + j) * BST + kr] = f2bf(wv);
      }
    }
    __syncthreads();

    v16bf a0 = load_frag(&As[(wm * 32 + lr)      * AST], half);
    v16bf a1 = load_frag(&As[(wm * 32 + 16 + lr) * AST], half);
    v16bf b0 = load_frag(&Bs[(wn * 32 + lr)      * BST], half);
    v16bf b1 = load_frag(&Bs[(wn * 32 + 16 + lr) * BST], half);
    acc[0][0] = wmma_bf16(a0, b0, acc[0][0]);
    acc[0][1] = wmma_bf16(a0, b1, acc[0][1]);
    acc[1][0] = wmma_bf16(a1, b0, acc[1][0]);
    acc[1][1] = wmma_bf16(a1, b1, acc[1][1]);
    __syncthreads();
  }

  // --- epilogue: D layout lane=col, VGPR r -> row r + 8*half ---
#pragma unroll
  for (int i = 0; i < 2; ++i) {
#pragma unroll
    for (int j = 0; j < 2; ++j) {
      int gn = n0 + wn * 32 + j * 16 + lr;
      float bv = (bias != nullptr && gn < N) ? bias[gn] : 0.f;
#pragma unroll
      for (int r = 0; r < 8; ++r) {
        int gm = m0 + wm * 32 + i * 16 + r + half * 8;
        if (gm < M && gn < N) {
          float v = acc[i][j][r] + bv;
          C[(size_t)gm * N + gn] = apply_act(v, act);
        }
      }
    }
  }
}

// ---------------------------------------------------------------------------
// NATTEN layer attention, fully in WMMA. One wave per (b, h, 16-query tile).
// Key window for query tile q0..q0+15 is [q0-16, q0+31] = 3 key tiles.
// Block = 64 threads = 2 waves. qkv: [8192, 1536] (q|k|v each 512 wide).
// ---------------------------------------------------------------------------
#define SQST 72
#define SKST 72
#define SVST 72
#define SPST 72
#define NAT_PER (16*SQST + 48*SKST + 64*SVST + 16*SPST)  // 10368 bf16 per wave

__global__ __launch_bounds__(64)
void natten_kernel(const float* __restrict__ qkv, float* __restrict__ attn)
{
  __shared__ __align__(16) bf16_t smem[2 * NAT_PER];
  const int w    = threadIdx.x >> 5;
  const int lane = threadIdx.x & 31;
  const int task = blockIdx.x * 2 + w;   // 0..4095
  const int qt   = task & 63;
  const int h    = (task >> 6) & 7;
  const int b    = task >> 9;
  const int half = lane >> 4;
  const int lr   = lane & 15;

  bf16_t* Sq = smem + w * NAT_PER;       // [16 q][hd 64]    (A for scores)
  bf16_t* Sk = Sq + 16 * SQST;           // [48 key][hd 64]  (B for scores)
  bf16_t* Sv = Sk + 48 * SKST;           // [hd 64][key 48+pad] (B for PV)
  bf16_t* Sp = Sv + 64 * SVST;           // [16 q][key 48+pad]  (A for PV)

  const int kk0 = qt * 16 - 16;
  const size_t rowq0 = (size_t)(b * 1024 + qt * 16);

  // stage Q: 16x64 -> bf16
  for (int id = lane; id < 256; id += 32) {
    int row = id >> 4, quad = id & 15;
    float4 v = *(const float4*)(qkv + (rowq0 + row) * 1536 + h * 64 + quad * 4);
    bf16_t p4[4] = {f2bf(v.x), f2bf(v.y), f2bf(v.z), f2bf(v.w)};
    *(uint2*)&Sq[row * SQST + quad * 4] = *(const uint2*)p4;
  }
  // stage K: 48x64, zero outside sequence
  for (int id = lane; id < 768; id += 32) {
    int row = id >> 4, quad = id & 15;
    int kk = kk0 + row;
    float4 v = make_float4(0.f, 0.f, 0.f, 0.f);
    if (kk >= 0 && kk < 1024)
      v = *(const float4*)(qkv + (size_t)(b * 1024 + kk) * 1536 + 512 + h * 64 + quad * 4);
    bf16_t p4[4] = {f2bf(v.x), f2bf(v.y), f2bf(v.z), f2bf(v.w)};
    *(uint2*)&Sk[row * SKST + quad * 4] = *(const uint2*)p4;
  }
  // stage V transposed: Sv[d][key]
  for (int id = lane; id < 768; id += 32) {
    int row = id >> 4, quad = id & 15;
    int kk = kk0 + row;
    float4 v = make_float4(0.f, 0.f, 0.f, 0.f);
    if (kk >= 0 && kk < 1024)
      v = *(const float4*)(qkv + (size_t)(b * 1024 + kk) * 1536 + 1024 + h * 64 + quad * 4);
    Sv[(quad * 4 + 0) * SVST + row] = f2bf(v.x);
    Sv[(quad * 4 + 1) * SVST + row] = f2bf(v.y);
    Sv[(quad * 4 + 2) * SVST + row] = f2bf(v.z);
    Sv[(quad * 4 + 3) * SVST + row] = f2bf(v.w);
  }
  // zero Sv key-pad cols 48..63 (read by K-chunk 1)
  for (int id = lane; id < 1024; id += 32) {
    int d = id >> 4, c = 48 + (id & 15);
    Sv[d * SVST + c] = f2bf(0.f);
  }
  __syncthreads();

  // scores: S[kt] = Q @ K^T   (3 key tiles x 2 K-chunks over hd=64)
  const v8f vz = {0.f, 0.f, 0.f, 0.f, 0.f, 0.f, 0.f, 0.f};
  v8f S[3]; S[0] = vz; S[1] = vz; S[2] = vz;
#pragma unroll
  for (int kc = 0; kc < 2; ++kc) {
    v16bf aq = load_frag(Sq + lr * SQST + kc * 32, half);
#pragma unroll
    for (int kt = 0; kt < 3; ++kt) {
      v16bf bk = load_frag(Sk + (kt * 16 + lr) * SKST + kc * 32, half);
      S[kt] = wmma_bf16(aq, bk, S[kt]);
    }
  }

  // mask + scale + per-row softmax (rows spread over VGPR r, cols over lanes)
  float sc[3][8];
#pragma unroll
  for (int kt = 0; kt < 3; ++kt) {
#pragma unroll
    for (int r = 0; r < 8; ++r) {
      int m_q = qt * 16 + r + half * 8;
      int key = kk0 + kt * 16 + lr;
      int dif = key - m_q; if (dif < 0) dif = -dif;
      bool valid = (key >= 0) && (key < 1024) && (dif <= 15);
      sc[kt][r] = valid ? S[kt][r] * 0.125f : -3.0e38f;
    }
  }
#pragma unroll
  for (int r = 0; r < 8; ++r) {
    float m = fmaxf(fmaxf(sc[0][r], sc[1][r]), sc[2][r]);
#pragma unroll
    for (int off = 1; off < 16; off <<= 1) m = fmaxf(m, __shfl_xor(m, off, 32));
    float e0 = __expf(sc[0][r] - m);
    float e1 = __expf(sc[1][r] - m);
    float e2 = __expf(sc[2][r] - m);
    float s = e0 + e1 + e2;
#pragma unroll
    for (int off = 1; off < 16; off <<= 1) s += __shfl_xor(s, off, 32);
    float inv = 1.f / s;
    int mrow = r + half * 8;
    Sp[mrow * SPST +  0 + lr] = f2bf(e0 * inv);
    Sp[mrow * SPST + 16 + lr] = f2bf(e1 * inv);
    Sp[mrow * SPST + 32 + lr] = f2bf(e2 * inv);
    Sp[mrow * SPST + 48 + lr] = f2bf(0.f);   // zero pad keys 48..63
  }
  __syncthreads();

  // out = P @ V : 2 K-chunks (48->64 padded keys) x 4 hd tiles
  v8f O[4]; O[0] = vz; O[1] = vz; O[2] = vz; O[3] = vz;
#pragma unroll
  for (int kc = 0; kc < 2; ++kc) {
    v16bf ap = load_frag(Sp + lr * SPST + kc * 32, half);
#pragma unroll
    for (int nt = 0; nt < 4; ++nt) {
      v16bf bv = load_frag(Sv + (nt * 16 + lr) * SVST + kc * 32, half);
      O[nt] = wmma_bf16(ap, bv, O[nt]);
    }
  }
#pragma unroll
  for (int nt = 0; nt < 4; ++nt) {
#pragma unroll
    for (int r = 0; r < 8; ++r) {
      int mrow = r + half * 8;
      attn[(rowq0 + mrow) * 512 + h * 64 + nt * 16 + lr] = O[nt][r];
    }
  }
}

// ---------------------------------------------------------------------------
// out = LN(a + resid) * g + b, optional clip to [-100, 100]. D = 512.
// One block (256 threads) per row.
// ---------------------------------------------------------------------------
__global__ __launch_bounds__(256)
void ln_residual(const float* __restrict__ a, const float* __restrict__ resid,
                 const float* __restrict__ g, const float* __restrict__ bta,
                 float* __restrict__ out, int clip)
{
  __shared__ float red[256];
  const int row = blockIdx.x, t = threadIdx.x;
  const float* pa = a + (size_t)row * 512;
  const float* pr = resid + (size_t)row * 512;
  float x0 = pa[t] + pr[t];
  float x1 = pa[t + 256] + pr[t + 256];

  red[t] = x0 + x1; __syncthreads();
  for (int o = 128; o > 0; o >>= 1) { if (t < o) red[t] += red[t + o]; __syncthreads(); }
  float mean = red[0] * (1.f / 512.f);
  __syncthreads();

  float d0 = x0 - mean, d1 = x1 - mean;
  red[t] = d0 * d0 + d1 * d1; __syncthreads();
  for (int o = 128; o > 0; o >>= 1) { if (t < o) red[t] += red[t + o]; __syncthreads(); }
  float rstd = rsqrtf(red[0] * (1.f / 512.f) + 1e-5f);

  float y0 = d0 * rstd * g[t] + bta[t];
  float y1 = d1 * rstd * g[t + 256] + bta[t + 256];
  if (clip) {
    y0 = fminf(fmaxf(y0, -100.f), 100.f);
    y1 = fminf(fmaxf(y1, -100.f), 100.f);
  }
  float* po = out + (size_t)row * 512;
  po[t] = y0; po[t + 256] = y1;
}

// mean over L: x[8,1024,512] -> meanx[8,512]. 16 blocks x 256 threads.
__global__ __launch_bounds__(256)
void mean_over_L(const float* __restrict__ x, float* __restrict__ meanx)
{
  int b = blockIdx.x >> 1;
  int col = (blockIdx.x & 1) * 256 + threadIdx.x;
  float s = 0.f;
  const float* p = x + ((size_t)b * 1024) * 512 + col;
  for (int l = 0; l < 1024; ++l) s += p[(size_t)l * 512];
  meanx[b * 512 + col] = s * (1.f / 1024.f);
}

// comb[row, c] += gb2[row/1024, c]
__global__ __launch_bounds__(256)
void add_broadcast(float* __restrict__ comb, const float* __restrict__ gb2)
{
  int idx = blockIdx.x * 256 + threadIdx.x;     // 8192*512 total
  int row = idx >> 9, c = idx & 511;
  comb[idx] += gb2[(row >> 10) * 512 + c];
}

// temb[row] = softmax(type_logits[row]) @ type_emb.  One thread per row.
__global__ __launch_bounds__(256)
void type_softmax_emb(const float* __restrict__ tlog, const float* __restrict__ emb,
                      float* __restrict__ temb)
{
  int row = blockIdx.x * 256 + threadIdx.x;     // 8192 rows (32 blocks)
  float lg[18];
  const float* tl = tlog + (size_t)row * 18;
  float m = -3.0e38f;
#pragma unroll
  for (int j = 0; j < 18; ++j) { lg[j] = tl[j]; m = fmaxf(m, lg[j]); }
  float s = 0.f;
#pragma unroll
  for (int j = 0; j < 18; ++j) { lg[j] = __expf(lg[j] - m); s += lg[j]; }
  float inv = 1.f / s;
  float* po = temb + (size_t)row * 32;
#pragma unroll
  for (int c = 0; c < 32; ++c) {
    float a = 0.f;
#pragma unroll
    for (int j = 0; j < 18; ++j) a += lg[j] * emb[j * 32 + c];
    po[c] = a * inv;
  }
}

// si = concat(enc[512], temb[32]) -> [8192, 544]
__global__ __launch_bounds__(256)
void concat_si(const float* __restrict__ enc, const float* __restrict__ temb,
               float* __restrict__ si)
{
  int idx = blockIdx.x * 256 + threadIdx.x;
  if (idx >= 8192 * 544) return;
  int row = idx / 544;
  int c = idx - row * 544;
  si[idx] = (c < 512) ? enc[(size_t)row * 512 + c] : temb[(size_t)row * 32 + (c - 512)];
}

// ---------------------------------------------------------------------------
// Host-side orchestration
// ---------------------------------------------------------------------------
static inline dim3 gemm_grid(int M, int N) { return dim3((N + 63) / 64, (M + 127) / 128); }

extern "C" void kernel_launch(void* const* d_in, const int* in_sizes, int n_in,
                              void* d_out, int out_size, void* d_ws, size_t ws_size,
                              hipStream_t stream) {
  (void)in_sizes; (void)n_in; (void)out_size; (void)ws_size;
  const float* state = (const float*)d_in[0];
  const float* Wp    = (const float*)d_in[1];
  const float* bp    = (const float*)d_in[2];
  const float* Wqkv  = (const float*)d_in[3];
  const float* bqkv  = (const float*)d_in[4];
  const float* Wo    = (const float*)d_in[5];
  const float* bo    = (const float*)d_in[6];
  const float* ln_g  = (const float*)d_in[7];
  const float* ln_b  = (const float*)d_in[8];
  const float* Wg    = (const float*)d_in[9];
  const float* bg    = (const float*)d_in[10];
  const float* Wc    = (const float*)d_in[11];
  const float* bc    = (const float*)d_in[12];
  const float* W1    = (const float*)d_in[13];
  const float* b1    = (const float*)d_in[14];
  const float* W2    = (const float*)d_in[15];
  const float* b2    = (const float*)d_in[16];
  const float* fn_g  = (const float*)d_in[17];
  const float* fn_b  = (const float*)d_in[18];
  const float* Wt1   = (const float*)d_in[19];
  const float* bt1   = (const float*)d_in[20];
  const float* Wt2   = (const float*)d_in[21];
  const float* bt2   = (const float*)d_in[22];
  const float* temb_w= (const float*)d_in[23];
  const float* Ws1   = (const float*)d_in[24];
  const float* bs1   = (const float*)d_in[25];
  const float* Ws2   = (const float*)d_in[26];
  const float* bs2   = (const float*)d_in[27];
  const float* Wa1   = (const float*)d_in[28];
  const float* ba1   = (const float*)d_in[29];
  const float* Wa2   = (const float*)d_in[30];
  const float* ba2   = (const float*)d_in[31];

  float* ws = (float*)d_ws;
  const size_t MEG = 1u << 20;
  float* x     = ws + 0;            // [0,4M)   ; later reused as enc
  float* qkv   = ws + 4 * MEG;      // [4M,16M) ; later comb(4M)/h1(8M)
  float* attn  = ws + 16 * MEG;     // [16M,20M); later ffn, then t1/s1
  float* locA  = ws + 20 * MEG;     // later si
  float* locB  = ws + 24 * MEG;
  float* proj  = ws + 4 * MEG;      // alias (qkv dead after natten)
  float* comb  = ws + 4 * MEG;
  float* h1    = ws + 8 * MEG;
  float* ffn   = ws + 16 * MEG;
  float* enc   = ws + 0;
  float* t1    = ws + 16 * MEG;
  float* s1    = ws + 18 * MEG;
  float* si    = ws + 20 * MEG;
  float* meanx = ws + 28 * MEG;
  float* gsum  = meanx + 4096;
  float* gb2   = meanx + 8192;
  float* temb  = meanx + 16384;

  const dim3 blk(256);

  // 1) x = state @ Wp + bp
  gemm_bias_act<<<gemm_grid(8192, 512), blk, 0, stream>>>(state, Wp, bp, x, 8192, 512, 256, 0);
  // 2) global branch (uses x before layers overwrite nothing, but compute now)
  mean_over_L<<<dim3(16), blk, 0, stream>>>(x, meanx);
  gemm_bias_act<<<gemm_grid(8, 512), blk, 0, stream>>>(meanx, Wg, bg, gsum, 8, 512, 512, 0);
  gemm_bias_act<<<gemm_grid(8, 512), blk, 0, stream>>>(gsum, Wc + 512 * 512, nullptr, gb2, 8, 512, 512, 0);

  // 3) two NATTEN layers
  const float* lin = x;
  float* louts[2] = {locA, locB};
  for (int i = 0; i < 2; ++i) {
    gemm_bias_act<<<gemm_grid(8192, 1536), blk, 0, stream>>>(
        lin, Wqkv + (size_t)i * 512 * 1536, bqkv + i * 1536, qkv, 8192, 1536, 512, 0);
    natten_kernel<<<dim3(2048), dim3(64), 0, stream>>>(qkv, attn);
    gemm_bias_act<<<gemm_grid(8192, 512), blk, 0, stream>>>(
        attn, Wo + (size_t)i * 512 * 512, bo + i * 512, proj, 8192, 512, 512, 0);
    ln_residual<<<dim3(8192), blk, 0, stream>>>(proj, lin, ln_g + i * 512, ln_b + i * 512, louts[i], 0);
    lin = louts[i];
  }

  // 4) combine + FFN + final LN(+clip)
  gemm_bias_act<<<gemm_grid(8192, 512), blk, 0, stream>>>(locB, Wc, bc, comb, 8192, 512, 512, 0);
  add_broadcast<<<dim3(16384), blk, 0, stream>>>(comb, gb2);
  gemm_bias_act<<<gemm_grid(8192, 1024), blk, 0, stream>>>(comb, W1, b1, h1, 8192, 1024, 512, 2);
  gemm_bias_act<<<gemm_grid(8192, 512), blk, 0, stream>>>(h1, W2, b2, ffn, 8192, 512, 1024, 0);
  ln_residual<<<dim3(8192), blk, 0, stream>>>(ffn, comb, fn_g, fn_b, enc, 1);

  // 5) heads
  float* out = (float*)d_out;
  float* type_logits   = out;
  float* size_logits   = out + 8192 * 18;
  float* amount_logits = out + 8192 * 18 + 8192 * 5;

  gemm_bias_act<<<gemm_grid(8192, 256), blk, 0, stream>>>(enc, Wt1, bt1, t1, 8192, 256, 512, 1);
  gemm_bias_act<<<gemm_grid(8192, 18), blk, 0, stream>>>(t1, Wt2, bt2, type_logits, 8192, 18, 256, 0);
  type_softmax_emb<<<dim3(32), blk, 0, stream>>>(type_logits, temb_w, temb);
  concat_si<<<dim3((8192 * 544 + 255) / 256), blk, 0, stream>>>(enc, temb, si);

  gemm_bias_act<<<gemm_grid(8192, 256), blk, 0, stream>>>(si, Ws1, bs1, s1, 8192, 256, 544, 1);
  gemm_bias_act<<<gemm_grid(8192, 5), blk, 0, stream>>>(s1, Ws2, bs2, size_logits, 8192, 5, 256, 0);
  gemm_bias_act<<<gemm_grid(8192, 256), blk, 0, stream>>>(si, Wa1, ba1, s1, 8192, 256, 544, 1);
  gemm_bias_act<<<gemm_grid(8192, 5), blk, 0, stream>>>(s1, Wa2, ba2, amount_logits, 8192, 5, 256, 0);
}
